// RAEScript_24455543783985
// MI455X (gfx1250) — compile-verified, matrix operand order checked by reading, not run on hardware
//
#include <hip/hip_runtime.h>

// ---------------------------------------------------------------------------
// RAE tree reduction for MI455X (gfx1250).
// Heap-layout complete binary tree => every level is a dense GEMM:
//   OUT[m,512] = tanh( Hchild[m,1024] @ W.T + b )
// bf16 WMMA (v_wmma_f32_16x16x32_bf16), f32 accumulate, H stored bf16.
// Double-buffered LDS fed by GLOBAL_LOAD_ASYNC_TO_LDS_B128 (ASYNCcnt),
// one barrier per K-step.
// ---------------------------------------------------------------------------

typedef __attribute__((ext_vector_type(16))) __bf16 v16bf;
typedef __attribute__((ext_vector_type(8)))  float  v8f;

union FragU { uint4 q[2]; v16bf v; };

__device__ __forceinline__ unsigned short f32_to_bf16(float f) {
    unsigned u = __float_as_uint(f);
    u += 0x7FFFu + ((u >> 16) & 1u);          // round-to-nearest-even
    return (unsigned short)(u >> 16);
}

// CDNA5 async copy: global -> LDS, 16 bytes per lane, tracked by ASYNCcnt.
__device__ __forceinline__ void async_load_b128(const void* gaddr, void* lds) {
    unsigned l = (unsigned)(size_t)lds;       // low 32 bits = LDS offset
    asm volatile("global_load_async_to_lds_b128 %0, %1, off"
                 :: "v"(l), "v"(gaddr) : "memory");
}
__device__ __forceinline__ void wait_asynccnt0() {
    asm volatile("s_wait_asynccnt 0x0" ::: "memory");
}

#define DHID 512
#define LDA  1024   // A row length (two concatenated children rows)

// --- W f32 -> bf16 one-shot conversion ------------------------------------
__global__ void wconv_kernel(const float* __restrict__ src,
                             unsigned short* __restrict__ dst, int n) {
    int i = blockIdx.x * blockDim.x + threadIdx.x;
    if (i < n) dst[i] = f32_to_bf16(src[i]);
}

// --- one tree level: [m,1024] x [1024,512] GEMM + bias + tanh -------------
// Block: 256 threads = 8 waves (2x4), tile 128(M) x 128(N), BK = 32.
// Each wave: 4 M-tiles x 2 N-tiles of 16x16 => 8 v8f accumulators.
template <bool A_IS_F32>
__global__ __launch_bounds__(256)
void rae_level_gemm(const void* __restrict__ Abase_v,
                    const unsigned short* __restrict__ Wb,   // [512,1024] bf16
                    const float* __restrict__ bias,          // [512] f32
                    unsigned short* __restrict__ Hb,         // bf16 internal H
                    int S, int m, float* __restrict__ out_root)
{
    __shared__ __align__(16) unsigned short ldsA[2][128 * 32];
    __shared__ __align__(16) unsigned short ldsB[2][128 * 32];

    const int tid  = threadIdx.x;
    const int lane = tid & 31;
    const int wave = tid >> 5;
    const int wm   = wave >> 2;            // 0..1  (M direction)
    const int wn   = wave & 3;             // 0..3  (N direction)
    const int m0   = blockIdx.y * 128;
    const int n0   = blockIdx.x * 128;

    const float*          Af = (const float*)Abase_v;
    const unsigned short* Ab = (const unsigned short*)Abase_v;

    // Pre-zero out-of-range A rows (partial tiles on small levels). These
    // rows are never targeted by an async load, so zeros persist in both
    // buffers for the whole K-loop.
    if (!A_IS_F32 && (m0 + 128 > m)) {
        for (int idx = tid; idx < 128 * 4; idx += 256) {
            int row = idx >> 2;
            int q   = idx & 3;
            if (m0 + row >= m) {
                uint4 z = {0u, 0u, 0u, 0u};
                *(uint4*)&ldsA[0][row * 32 + q * 8] = z;
                *(uint4*)&ldsA[1][row * 32 + q * 8] = z;
            }
        }
    }

    // Stage one 128x32 A tile + 128x32 B tile into LDS buffer `buf`.
    auto stage = [&](int k0s, int buf) {
        for (int idx = tid; idx < 128 * 4; idx += 256) {
            int row = idx >> 2;
            int q   = idx & 3;
            // B tile: W rows n0..n0+127, cols k0s..k0s+31 (bf16, in-range)
            async_load_b128(Wb + (size_t)(n0 + row) * LDA + k0s + q * 8,
                            &ldsB[buf][row * 32 + q * 8]);
            if (A_IS_F32) {
                // leaf level: convert f32 children rows to bf16 on the fly
                // (m = 32768 here, tiles always full -> no guard)
                size_t off = (size_t)(m0 + row) * LDA + (size_t)k0s + (size_t)q * 8;
                float4 a0 = *(const float4*)(Af + off);
                float4 a1 = *(const float4*)(Af + off + 4);
                uint4 v;
                v.x = (unsigned)f32_to_bf16(a0.x) | ((unsigned)f32_to_bf16(a0.y) << 16);
                v.y = (unsigned)f32_to_bf16(a0.z) | ((unsigned)f32_to_bf16(a0.w) << 16);
                v.z = (unsigned)f32_to_bf16(a1.x) | ((unsigned)f32_to_bf16(a1.y) << 16);
                v.w = (unsigned)f32_to_bf16(a1.z) | ((unsigned)f32_to_bf16(a1.w) << 16);
                *(uint4*)&ldsA[buf][row * 32 + q * 8] = v;
            } else {
                int grow = m0 + row;
                if (grow < m)
                    async_load_b128(Ab + (size_t)grow * LDA + k0s + q * 8,
                                    &ldsA[buf][row * 32 + q * 8]);
            }
        }
    };

    v8f acc[4][2] = {};

    stage(0, 0);                            // prologue: fill buffer 0

    const int r  = lane & 15;
    const int kb = (lane >> 4) * 8;

    for (int kt = 0; kt < LDA / 32; ++kt) {
        const int cur = kt & 1;
        wait_asynccnt0();                   // this wave's async fills done
        __syncthreads();                    // everyone's fills visible
        if (kt + 1 < LDA / 32)
            stage((kt + 1) * 32, cur ^ 1);  // overlap next fills with WMMA

        // Fragments (ISA 16-bit A layout: lanes 0-15 K0-7 & 16-23,
        // lanes 16-31 K8-15 & 24-31; B mirrored per-column).
        FragU afr[4], bfr[2];
        #pragma unroll
        for (int t = 0; t < 4; ++t) {
            int arow = wm * 64 + t * 16 + r;
            afr[t].q[0] = *(const uint4*)&ldsA[cur][arow * 32 + kb];
            afr[t].q[1] = *(const uint4*)&ldsA[cur][arow * 32 + kb + 16];
        }
        #pragma unroll
        for (int u = 0; u < 2; ++u) {
            int brow = wn * 32 + u * 16 + r;
            bfr[u].q[0] = *(const uint4*)&ldsB[cur][brow * 32 + kb];
            bfr[u].q[1] = *(const uint4*)&ldsB[cur][brow * 32 + kb + 16];
        }
        #pragma unroll
        for (int t = 0; t < 4; ++t)
            #pragma unroll
            for (int u = 0; u < 2; ++u)
                acc[t][u] = __builtin_amdgcn_wmma_f32_16x16x32_bf16(
                    false, afr[t].v, false, bfr[u].v,
                    (short)0, acc[t][u], false, false);
    }

    // ---- epilogue: bias + tanh, store bf16 H (and f32 root row) -----------
    const int nl = lane & 15;
    const int mh = (lane >> 4) * 8;
    #pragma unroll
    for (int u = 0; u < 2; ++u) {
        int gn = n0 + wn * 32 + u * 16 + nl;
        float bv = bias[gn];
        #pragma unroll
        for (int t = 0; t < 4; ++t) {
            #pragma unroll
            for (int v = 0; v < 8; ++v) {
                int gm = m0 + wm * 64 + t * 16 + mh + v;
                if (gm < m) {
                    float val = tanhf(acc[t][u][v] + bv);
                    Hb[(size_t)(S + gm) * DHID + gn] = f32_to_bf16(val);
                    if (out_root != nullptr && gm == 0)
                        out_root[gn] = val;
                }
            }
        }
    }
}

// ---------------------------------------------------------------------------
// inputs: [0]=left(i64) [1]=right(i64) [2]=is_leaf(i64) [3]=inp f32[131071,512]
//         [4]=root(scalar) [5]=W f32[512,1024] [6]=b f32[512]
// output: H[root] -> 512 f32
// workspace: Wb bf16 [512*1024] (1 MB) + Hb bf16 [65535*512] (~67 MB)
// ---------------------------------------------------------------------------
extern "C" void kernel_launch(void* const* d_in, const int* in_sizes, int n_in,
                              void* d_out, int out_size, void* d_ws, size_t ws_size,
                              hipStream_t stream)
{
    const float* inp = (const float*)d_in[3];
    const float* W   = (const float*)d_in[5];
    const float* b   = (const float*)d_in[6];
    float* out       = (float*)d_out;

    unsigned short* Wb = (unsigned short*)d_ws;
    unsigned short* Hb = (unsigned short*)((char*)d_ws +
                          (size_t)512 * 1024 * sizeof(unsigned short));

    wconv_kernel<<<(512 * 1024 + 255) / 256, 256, 0, stream>>>(W, Wb, 512 * 1024);

    for (int lvl = 15; lvl >= 0; --lvl) {
        int m = 1 << lvl;
        int S = m - 1;                       // first node of this level
        dim3 grid(DHID / 128, (m + 127) / 128);
        float* oroot = (lvl == 0) ? out : nullptr;
        // children of node S start at 2S+1; rows are contiguous pairs
        if (lvl == 15) {
            rae_level_gemm<true><<<grid, 256, 0, stream>>>(
                (const void*)(inp + (size_t)(2 * S + 1) * DHID),
                Wb, b, Hb, S, m, oroot);
        } else {
            rae_level_gemm<false><<<grid, 256, 0, stream>>>(
                (const void*)(Hb + (size_t)(2 * S + 1) * DHID),
                Wb, b, Hb, S, m, oroot);
        }
    }
}
